// SGLinearAttention_25658134626478
// MI455X (gfx1250) — matrix operand order repaired and checked by reading, not compiled, not run on hardware
//
#include <hip/hip_runtime.h>

typedef __bf16 bf16;
typedef __attribute__((ext_vector_type(16))) __bf16 v16bf;
typedef __attribute__((ext_vector_type(8)))  __bf16 v8bf;
typedef __attribute__((ext_vector_type(4)))  __bf16 v4bf;
typedef __attribute__((ext_vector_type(8)))  float  v8f;
typedef __attribute__((ext_vector_type(4)))  int    v4i;

#define DIM   256
#define TILE  128
#define LDA   264            // padded LDS row stride (bf16 elems): 528 B = 33*16 -> keeps 16B alignment
#define NT    16             // 256/16 n-tiles
#define KS    8              // 256/32 k-steps

#if __has_builtin(__builtin_amdgcn_global_load_async_to_lds_b128)
#define HAVE_ASYNC_LDS 1
#else
#define HAVE_ASYNC_LDS 0
#endif

__device__ __forceinline__ v8f wmma_bf16(v16bf a, v16bf b, v8f c) {
  return __builtin_amdgcn_wmma_f32_16x16x32_bf16(false, a, false, b, (short)0, c,
                                                 false, false);
}

// Swizzled B layout for a row-major [256][256] matrix:
// element (k, n) -> [kstep=k>>5][ntile=n>>4][lane=(k&31>>4)*16+(n&15)][h=k&15]
// so a 16x32 B-fragment for (kstep, ntile) is 32 contiguous bytes per lane.
__device__ __forceinline__ int bswz_idx(int k, int n) {
  int kstep = k >> 5, kk = k & 31;
  return (((kstep * 16 + (n >> 4)) * 32) + ((kk >> 4) << 4) + (n & 15)) * 16 + (kk & 15);
}

__device__ __forceinline__ v16bf load_b_frag(const bf16* __restrict__ wsw, int kstep,
                                             int ntile, int lane) {
  return *(const v16bf*)(wsw + ((kstep * 16 + ntile) * 32 + lane) * 16);  // 2x b128
}

// Load a 16x32 bf16 A-fragment from row-major padded LDS (base = row 0 of strip).
// Layout: lanes 0-15 hold M=lane, K = {0..15}; lanes 16-31 K = {16..31}.
__device__ __forceinline__ v16bf load_a_frag(const bf16* base, int k0, int lane) {
  int m = lane & 15, hs = lane >> 4;
  const bf16* p = base + m * LDA + k0 + hs * 16;
  v8bf lo = *(const v8bf*)p;
  v8bf hi = *(const v8bf*)(p + 8);
  v16bf a;
#pragma unroll
  for (int h = 0; h < 8; ++h) { a[h] = lo[h]; a[8 + h] = hi[h]; }
  return a;
}

// 128x256 tile GEMM: acc[n] = bias + x_tile @ W  (A-frags in regs, B streamed from L2)
__device__ __forceinline__ void gemm_tile(const v16bf* aX, const bf16* __restrict__ wsw,
                                          const float* __restrict__ bias, int lane,
                                          v8f* acc) {
#pragma unroll
  for (int n = 0; n < NT; ++n) {
    float bv_ = bias[n * 16 + (lane & 15)];
#pragma unroll
    for (int r = 0; r < 8; ++r) acc[n][r] = bv_;
  }
#pragma unroll
  for (int k = 0; k < KS; ++k) {
    if (k + 1 < KS)
      __builtin_prefetch(wsw + ((k + 1) * 16 * 32) * 16 + lane * 64, 0, 1);
#pragma unroll
    for (int n = 0; n < NT; ++n)
      acc[n] = wmma_bf16(aX[k], load_b_frag(wsw, k, n, lane), acc[n]);
  }
}

__global__ void k_init(const float* __restrict__ Wq, const float* __restrict__ Wk,
                       const float* __restrict__ Wv, bf16* __restrict__ Wqb,
                       bf16* __restrict__ Wkb, bf16* __restrict__ Wvb,
                       float* __restrict__ KV, float* __restrict__ ksum) {
  int i = blockIdx.x * blockDim.x + threadIdx.x;   // 65536 threads
  int idx = bswz_idx(i >> 8, i & 255);
  Wqb[idx] = (bf16)Wq[i];
  Wkb[idx] = (bf16)Wk[i];
  Wvb[idx] = (bf16)Wv[i];
  KV[i] = 0.0f;
  if (i < DIM) ksum[i] = 0.0f;
}

__global__ void k_kv2bf(const float* __restrict__ KV, bf16* __restrict__ KVb) {
  int i = blockIdx.x * blockDim.x + threadIdx.x;   // KV element (d = k-dim, e = n)
  KVb[bswz_idx(i >> 8, i & 255)] = (bf16)KV[i];
}

__global__ __launch_bounds__(256)
void k_pass1(const float* __restrict__ x,
             const bf16* __restrict__ Wqb, const float* __restrict__ bq,
             const bf16* __restrict__ Wkb, const float* __restrict__ bk,
             const bf16* __restrict__ Wvb, const float* __restrict__ bv,
             bf16* __restrict__ Qout, float* __restrict__ KV,
             float* __restrict__ ksum, int N) {
  extern __shared__ char smem[];
  bf16*  sX   = (bf16*)smem;                       // TILE*LDA (x tile; dead after A-frag cache)
  bf16*  sVsw = (bf16*)smem;                       // aliases sX: 32768 halves, B-swizzled V
  bf16*  sKsw = (bf16*)(smem + TILE * LDA * 2);    // 32768 halves, A-swizzled K^T
  float* sCS  = (float*)(smem + TILE * LDA * 2 + 65536); // 256 column sums

  int tid  = threadIdx.x;
  int lane = tid & 31, wave = tid >> 5;
  int m = lane & 15, hs = lane >> 4;
  int row0 = blockIdx.x * TILE;

  // ---- stage x tile -> LDS bf16 (zero-fill OOB rows) ----
#pragma unroll
  for (int i = 0; i < 32; ++i) {
    int e4 = tid + i * 256;            // 8192 float4 chunks
    int r = e4 >> 6, c4 = e4 & 63;
    int grow = row0 + r;
    float4 v = make_float4(0.f, 0.f, 0.f, 0.f);
    if (grow < N) v = *(const float4*)(x + grow * DIM + c4 * 4);
    v4bf o; o[0] = (bf16)v.x; o[1] = (bf16)v.y; o[2] = (bf16)v.z; o[3] = (bf16)v.w;
    *(v4bf*)(sX + r * LDA + c4 * 4) = o;
  }
  sCS[tid] = 0.0f;
  __syncthreads();

  // ---- cache this wave's 16x256 A-fragments in registers ----
  const bf16* aBase = sX + (wave * 16) * LDA;
  v16bf aX[KS];
#pragma unroll
  for (int k = 0; k < KS; ++k) aX[k] = load_a_frag(aBase, k * 32, lane);
  __syncthreads();   // all waves cached -> sX reusable as sVsw

  v8f acc[NT];

  // ================= Q = relu(x@Wq + bq) =================
  gemm_tile(aX, Wqb, bq, lane, acc);
#pragma unroll
  for (int n = 0; n < NT; ++n)
#pragma unroll
    for (int r = 0; r < 8; ++r) {
      int R = wave * 16 + hs * 8 + r;
      int grow = row0 + R;
      if (grow < N) Qout[grow * DIM + n * 16 + m] = (bf16)fmaxf(acc[n][r], 0.0f);
    }

  // ================= K = relu(x@Wk + bk) =================
  gemm_tile(aX, Wkb, bk, lane, acc);
#pragma unroll
  for (int n = 0; n < NT; ++n) {
    float csum = 0.0f;
#pragma unroll
    for (int r = 0; r < 8; ++r) {
      int R = wave * 16 + hs * 8 + r;
      int grow = row0 + R;
      float kv_ = (grow < N) ? fmaxf(acc[n][r], 0.0f) : 0.0f;
      csum += kv_;
      // write K^T into A-fragment-swizzled layout: A[m=d][kk=local row]
      int kstep = R >> 5, kk32 = R & 31;
      int hs2 = kk32 >> 4;
      int h2  = kk32 & 15;
      sKsw[((kstep * 16 + n) * 32 + hs2 * 16 + m) * 16 + h2] = (bf16)kv_;
    }
    atomicAdd(&sCS[n * 16 + m], csum);   // ds_add_f32
  }

  // ================= V = x@Wv + bv =================
  gemm_tile(aX, Wvb, bv, lane, acc);
#pragma unroll
  for (int n = 0; n < NT; ++n)
#pragma unroll
    for (int r = 0; r < 8; ++r) {
      int R = wave * 16 + hs * 8 + r;
      int grow = row0 + R;
      float vv = (grow < N) ? acc[n][r] : 0.0f;
      // write V into B-fragment-swizzled layout: B[kk=local row][n=col]
      int kstep = R >> 5;
      sVsw[((kstep * 16 + n) * 32 + ((R >> 4) & 1) * 16 + m) * 16 + (R & 15)] = (bf16)vv;
    }
  __syncthreads();

  // ---- flush column sums: ksum += sum_n K ----
  atomicAdd(&ksum[tid], sCS[tid]);       // global_atomic_add_f32

  // ---- KV += K^T @ V (this block's 128-row contribution) ----
#pragma unroll
  for (int dsel = 0; dsel < 2; ++dsel) {
    int dstrip = wave + dsel * 8, d0 = dstrip * 16;
    v16bf aK[4];
#pragma unroll
    for (int k = 0; k < 4; ++k)
      aK[k] = *(const v16bf*)(sKsw + ((k * 16 + dstrip) * 32 + lane) * 16);
#pragma unroll
    for (int n = 0; n < NT; ++n) {
      v8f c;
#pragma unroll
      for (int r = 0; r < 8; ++r) c[r] = 0.0f;
#pragma unroll
      for (int k = 0; k < 4; ++k)
        c = wmma_bf16(aK[k], *(const v16bf*)(sVsw + ((k * 16 + n) * 32 + lane) * 16), c);
#pragma unroll
      for (int r = 0; r < 8; ++r)
        atomicAdd(&KV[(d0 + hs * 8 + r) * DIM + n * 16 + m], c[r]);
    }
  }
}

__global__ __launch_bounds__(256)
void k_pass2(const bf16* __restrict__ Qb, const bf16* __restrict__ KVb,
             const float* __restrict__ ksum, float* __restrict__ out, int N) {
  extern __shared__ char smem[];
  bf16*  sQ    = (bf16*)smem;                          // TILE*LDA
  float* sNorm = (float*)(smem + TILE * LDA * 2);      // 128 f32

  int tid = threadIdx.x, lane = tid & 31, wave = tid >> 5;
  int m = lane & 15, hs = lane >> 4;
  int row0 = blockIdx.x * TILE;

  // ---- stage Q tile (bf16, 16B chunks); async DMA to LDS when available ----
#pragma unroll
  for (int i = 0; i < 16; ++i) {
    int e8 = tid + i * 256;             // 4096 chunks of 8 bf16
    int r = e8 >> 5, c8 = e8 & 31;
    int grow = row0 + r;
    bf16* dst = sQ + r * LDA + c8 * 8;
    if (grow < N) {
#if HAVE_ASYNC_LDS
      __builtin_amdgcn_global_load_async_to_lds_b128(
          (v4i*)(Qb + grow * DIM + c8 * 8), (v4i*)dst, 0, 0);
#else
      *(v8bf*)dst = *(const v8bf*)(Qb + grow * DIM + c8 * 8);
#endif
    } else {
      v8bf z;
#pragma unroll
      for (int j = 0; j < 8; ++j) z[j] = (bf16)0.0f;
      *(v8bf*)dst = z;
    }
  }
#if HAVE_ASYNC_LDS
#if __has_builtin(__builtin_amdgcn_s_wait_asynccnt)
  __builtin_amdgcn_s_wait_asynccnt(0);
#else
  asm volatile("s_wait_asynccnt 0x0" ::: "memory");
#endif
#endif
  __syncthreads();

  // ---- per-row normalizer: 1 / (Q . ksum + eps) ----
  if (tid < TILE) {
    float s = 0.0f;
    for (int c = 0; c < DIM; ++c) s += (float)sQ[tid * LDA + c] * ksum[c];
    sNorm[tid] = 1.0f / (s + 1e-6f);
  }

  // ---- out = Q @ KV (KV bf16, fragment-swizzled, streamed from L2) ----
  const bf16* aBase = sQ + (wave * 16) * LDA;
  v16bf aQ[KS];
#pragma unroll
  for (int k = 0; k < KS; ++k) aQ[k] = load_a_frag(aBase, k * 32, lane);

  v8f acc[NT];
#pragma unroll
  for (int n = 0; n < NT; ++n)
#pragma unroll
    for (int r = 0; r < 8; ++r) acc[n][r] = 0.0f;

#pragma unroll
  for (int k = 0; k < KS; ++k) {
    if (k + 1 < KS)
      __builtin_prefetch(KVb + ((k + 1) * 16 * 32) * 16 + lane * 64, 0, 1);
#pragma unroll
    for (int n = 0; n < NT; ++n)
      acc[n] = wmma_bf16(aQ[k], load_b_frag(KVb, k, n, lane), acc[n]);
  }
  __syncthreads();

#pragma unroll
  for (int n = 0; n < NT; ++n)
#pragma unroll
    for (int r = 0; r < 8; ++r) {
      int R = wave * 16 + hs * 8 + r;
      int grow = row0 + R;
      if (grow < N) out[grow * DIM + n * 16 + m] = acc[n][r] * sNorm[R];
    }
}

extern "C" void kernel_launch(void* const* d_in, const int* in_sizes, int n_in,
                              void* d_out, int out_size, void* d_ws, size_t ws_size,
                              hipStream_t stream) {
  const float* x  = (const float*)d_in[0];
  const float* Wq = (const float*)d_in[1];
  const float* bq = (const float*)d_in[2];
  const float* Wk = (const float*)d_in[3];
  const float* bk = (const float*)d_in[4];
  const float* Wv = (const float*)d_in[5];
  const float* bv = (const float*)d_in[6];
  const int N = in_sizes[0] / DIM;

  // workspace layout (bytes)
  char* ws = (char*)d_ws;
  bf16*  Wqb  = (bf16*)(ws + 0);          // 131072, B-fragment swizzled
  bf16*  Wkb  = (bf16*)(ws + 131072);     // 131072, B-fragment swizzled
  bf16*  Wvb  = (bf16*)(ws + 262144);     // 131072, B-fragment swizzled
  float* KV   = (float*)(ws + 393216);    // 262144 (f32 accumulator)
  bf16*  KVb  = (bf16*)(ws + 655360);     // 131072, B-fragment swizzled
  float* ksum = (float*)(ws + 786432);    // 1024
  bf16*  Qb   = (bf16*)(ws + 787456);     // N*512 bytes (bf16 Q)

  const int nblk = (N + TILE - 1) / TILE;
  const size_t lds1 = (size_t)TILE * LDA * 2 + 65536 + 1024;  // 134144 B
  const size_t lds2 = (size_t)TILE * LDA * 2 + 512;           // 68096 B

  k_init<<<256, 256, 0, stream>>>(Wq, Wk, Wv, Wqb, Wkb, Wvb, KV, ksum);
  k_pass1<<<nblk, 256, lds1, stream>>>(x, Wqb, bq, Wkb, bk, Wvb, bv, Qb, KV, ksum, N);
  k_kv2bf<<<256, 256, 0, stream>>>(KV, KVb);
  k_pass2<<<nblk, 256, lds2, stream>>>(Qb, KVb, ksum, (float*)d_out, N);
}
